// CGCNN_66236985639293
// MI455X (gfx1250) — matrix operand order repaired and checked by reading, not compiled, not run on hardware
//
#include <hip/hip_runtime.h>
#include <hip/hip_bf16.h>
#include <math.h>

// ---------------- problem constants (from reference) ----------------
#define N_NODES 50000
#define N_EDGES 800000
#define N_GRAPH 256
#define NDIM    3
#define EDIM    32
#define HDIM    128
#define CLAMP_V 1.0e6f

// edge-GEMM tiling
#define TILE_E  16            // edges per wave-tile (WMMA M)
#define KTILES  9             // 288 / 32
#define NTILES  8             // 128 / 16
#define ZROW    296           // 288 halfs + 8 pad (row stride 592B, 16B multiple)
#define WAVES   4             // waves per block in edge kernel

typedef __attribute__((ext_vector_type(16))) _Float16 v16h;
typedef __attribute__((ext_vector_type(8)))  float    v8f;

__device__ __forceinline__ float sigmoidf_(float x) { return 1.f / (1.f + __expf(-x)); }
__device__ __forceinline__ float softplusf_(float x) {
  return fmaxf(x, 0.f) + __logf(1.f + __expf(-fabsf(x)));
}

// ---------------- small helper kernels ----------------
__global__ __launch_bounds__(256) void k_copy_f32(float* __restrict__ dst,
                                                  const float* __restrict__ src, int n) {
  int i = blockIdx.x * blockDim.x + threadIdx.x;
  if (i < n) dst[i] = src[i];
}

__global__ __launch_bounds__(256) void k_f32_to_f16(const float* __restrict__ a,
                                                    _Float16* __restrict__ o, int n) {
  int i = blockIdx.x * blockDim.x + threadIdx.x;
  if (i < n) o[i] = (_Float16)a[i];
}

// pack W[288x128] f32 row-major into per-fragment WMMA B register order (f16):
// frag = kt*NTILES + nt ; per lane 16 halfs, K = kt*32 + 8*(lane>>4) + (p&7) + 16*(p>>3), col = nt*16 + (lane&15)
__global__ __launch_bounds__(256) void k_pack_w(const float* __restrict__ W,
                                                _Float16* __restrict__ out) {
  int t = blockIdx.x * blockDim.x + threadIdx.x;
  if (t >= KTILES * NTILES * 32) return;
  int lane = t & 31;
  int frag = t >> 5;
  int kt = frag / NTILES, nt = frag % NTILES;
  int g = lane >> 4, n = lane & 15;
  int col = nt * 16 + n;
  _Float16* o = out + (size_t)t * 16;
#pragma unroll
  for (int p = 0; p < 16; ++p)
    o[p] = (_Float16)W[(size_t)(kt * 32 + 8 * g + (p & 7) + 16 * (p >> 3)) * HDIM + col];
}

// conv1: tiny 38->3 per-edge MLP gate, scatter-add into h1 (pre-initialized to x)
__global__ __launch_bounds__(256) void k_conv1(const float* __restrict__ x,
                                               const int* __restrict__ ei,
                                               const float* __restrict__ ea,
                                               const float* __restrict__ Wf,
                                               const float* __restrict__ bf,
                                               const float* __restrict__ Ws,
                                               const float* __restrict__ bs,
                                               float* __restrict__ h1) {
  int e = blockIdx.x * blockDim.x + threadIdx.x;
  if (e >= N_EDGES) return;
  int s = ei[e];
  int d = ei[N_EDGES + e];
  float z[38];
#pragma unroll
  for (int k = 0; k < 3; ++k) z[k] = x[d * 3 + k];
#pragma unroll
  for (int k = 0; k < 3; ++k) z[3 + k] = x[s * 3 + k];
#pragma unroll
  for (int k = 0; k < 32; ++k) z[6 + k] = ea[(size_t)e * 32 + k];
#pragma unroll
  for (int j = 0; j < 3; ++j) {
    float f = bf[j], t = bs[j];
#pragma unroll
    for (int k = 0; k < 38; ++k) {
      f += z[k] * Wf[k * 3 + j];
      t += z[k] * Ws[k * 3 + j];
    }
    atomicAdd(&h1[d * 3 + j], sigmoidf_(f) * softplusf_(t));
  }
}

// proj(3->128) + layernorm + relu ; one block (128 thr) per node
__global__ __launch_bounds__(HDIM) void k_proj_ln(const float* __restrict__ h1,
                                                  const float* __restrict__ Wp,
                                                  const float* __restrict__ bp,
                                                  const float* __restrict__ ng,
                                                  const float* __restrict__ nb,
                                                  float* __restrict__ hcur,
                                                  _Float16* __restrict__ h16) {
  __shared__ float red[HDIM];
  int n = blockIdx.x, j = threadIdx.x;
  float a0 = h1[n * 3 + 0], a1 = h1[n * 3 + 1], a2 = h1[n * 3 + 2];
  float v = bp[j] + a0 * Wp[j] + a1 * Wp[HDIM + j] + a2 * Wp[2 * HDIM + j];
  red[j] = v; __syncthreads();
  for (int off = HDIM / 2; off > 0; off >>= 1) {
    if (j < off) red[j] += red[j + off];
    __syncthreads();
  }
  float mean = red[0] * (1.f / HDIM); __syncthreads();
  float dvi = v - mean;
  red[j] = dvi * dvi; __syncthreads();
  for (int off = HDIM / 2; off > 0; off >>= 1) {
    if (j < off) red[j] += red[j + off];
    __syncthreads();
  }
  float var = red[0] * (1.f / HDIM);
  float o = dvi * rsqrtf(var + 1e-5f) * ng[j] + nb[j];
  o = fmaxf(o, 0.f);
  hcur[(size_t)n * HDIM + j] = o;
  h16[(size_t)n * HDIM + j] = (_Float16)o;
}

// relu(clip(acc)) -> hcur (f32) and h16 (f16)
__global__ __launch_bounds__(256) void k_post(const float* __restrict__ acc,
                                              float* __restrict__ hcur,
                                              _Float16* __restrict__ h16, int n) {
  int i = blockIdx.x * blockDim.x + threadIdx.x;
  if (i >= n) return;
  float v = acc[i];
  v = fmaxf(0.f, fminf(v, CLAMP_V));
  hcur[i] = v;
  h16[i] = (_Float16)v;
}

// ---------------- the WMMA edge-GEMM kernel (conv2/conv3) ----------------
// Per wave: gather 16 z rows (288 f16) into private LDS, then for each of 8 n-tiles
// run 9 K-steps of v_wmma_f32_16x16x32_f16 against pre-packed Wf and Ws fragments,
// gate, and atomic-scatter to acc[dst].
__global__ __launch_bounds__(WAVES * 32)
void k_edge_wmma(const _Float16* __restrict__ h16, const _Float16* __restrict__ ea16,
                 const int* __restrict__ ei,
                 const _Float16* __restrict__ wpF, const _Float16* __restrict__ wpS,
                 const float* __restrict__ bF, const float* __restrict__ bS,
                 float* __restrict__ acc, int numTiles) {
  __shared__ _Float16 zbuf[WAVES][TILE_E][ZROW];
  __shared__ int dnode[WAVES][TILE_E];
  const int wave = threadIdx.x >> 5;
  const int lane = threadIdx.x & 31;
  const int g = lane >> 4;     // lane group (selects K half per WMMA layout)
  const int nlo = lane & 15;
  const int totWaves = gridDim.x * WAVES;

  for (int tile = blockIdx.x * WAVES + wave; tile < numTiles; tile += totWaves) {
    const int eBase = tile * TILE_E;
    // ---- gather z tile: 36 chunks of 8 halfs per row, b128 moves ----
    for (int r = 0; r < TILE_E; ++r) {
      const int e = eBase + r;
      const int s = ei[e];
      const int d = ei[N_EDGES + e];
      if (lane == 0) dnode[wave][r] = d;
      const int c = lane;
      const _Float16* src =
          (c < 16) ? (h16 + (size_t)d * HDIM + c * 8)
        : (c < 32) ? (h16 + (size_t)s * HDIM + (c - 16) * 8)
                   : (ea16 + (size_t)e * EDIM + (c - 32) * 8);
      *(uint4*)&zbuf[wave][r][c * 8] = *(const uint4*)src;
      if (lane < 4) {
        const int c2 = 32 + lane;
        *(uint4*)&zbuf[wave][r][c2 * 8] =
            *(const uint4*)(ea16 + (size_t)e * EDIM + (c2 - 32) * 8);
      }
    }
    // wave-private LDS region; DS ops are in-order within a wave -> no barrier.

    // A row for this lane: m = lane&15 (both lane groups), K pairs per ISA A layout.
    const unsigned int* zr32 = (const unsigned int*)&zbuf[wave][nlo][0];

    for (int nt = 0; nt < NTILES; ++nt) {
      union { v8f v; float f[8]; } accF, accS;
#pragma unroll
      for (int q = 0; q < 8; ++q) { accF.f[q] = 0.f; accS.f[q] = 0.f; }

#pragma unroll
      for (int kt = 0; kt < KTILES; ++kt) {
        union { unsigned int u[8]; v16h v; } a;
        const int base = (kt * 32 + 8 * g) >> 1;  // u32 index into z row
#pragma unroll
        for (int q = 0; q < 4; ++q) {
          a.u[q]     = zr32[base + q];        // K = 8g + {0..7}
          a.u[4 + q] = zr32[base + 8 + q];    // K = 16 + 8g + {0..7}
        }
        const size_t wi = ((size_t)(kt * NTILES + nt) * 32 + lane) * 16;
        v16h bfm = *(const v16h*)(wpF + wi);
        v16h bsm = *(const v16h*)(wpS + wi);
        accF.v = __builtin_amdgcn_wmma_f32_16x16x32_f16(false, a.v, false, bfm,
                                                        (short)0, accF.v, false, false);
        accS.v = __builtin_amdgcn_wmma_f32_16x16x32_f16(false, a.v, false, bsm,
                                                        (short)0, accS.v, false, false);
      }

      const int col = nt * 16 + nlo;
      const float bfc = bF[col];
      const float bsc = bS[col];
#pragma unroll
      for (int r = 0; r < 8; ++r) {
        const int m = r + 8 * g;              // C layout: vgpr r, lanes16-31 -> M=r+8
        const float msg = sigmoidf_(accF.f[r] + bfc) * softplusf_(accS.f[r] + bsc);
        atomicAdd(&acc[(size_t)dnode[wave][m] * HDIM + col], msg);
      }
    }
  }
}

// ---------------- pooling + head ----------------
__global__ __launch_bounds__(256) void k_pool_cnt(const int* __restrict__ batch,
                                                  float* __restrict__ cnt) {
  int n = blockIdx.x * blockDim.x + threadIdx.x;
  if (n < N_NODES) atomicAdd(&cnt[batch[n]], 1.f);
}

__global__ __launch_bounds__(256) void k_pool_sum(const float* __restrict__ hcur,
                                                  const int* __restrict__ batch,
                                                  float* __restrict__ pooled) {
  int i = blockIdx.x * blockDim.x + threadIdx.x;
  if (i >= N_NODES * HDIM) return;
  int n = i >> 7, c = i & (HDIM - 1);
  atomicAdd(&pooled[(size_t)batch[n] * HDIM + c], hcur[i]);
}

__global__ __launch_bounds__(HDIM) void k_final(const float* __restrict__ pooled,
                                                const float* __restrict__ cnt,
                                                const float* __restrict__ W1,
                                                const float* __restrict__ b1,
                                                const float* __restrict__ ng,
                                                const float* __restrict__ nb,
                                                const float* __restrict__ Wh,
                                                const float* __restrict__ bh,
                                                float* __restrict__ out) {
  __shared__ float p[HDIM], t[HDIM], red[HDIM];
  int gidx = blockIdx.x, j = threadIdx.x;
  float c = fmaxf(cnt[gidx], 1.f);
  p[j] = pooled[(size_t)gidx * HDIM + j] / c;
  __syncthreads();
  float v = b1[j];
  for (int k = 0; k < HDIM; ++k) v += p[k] * W1[k * HDIM + j];
  red[j] = v; __syncthreads();
  for (int off = HDIM / 2; off > 0; off >>= 1) {
    if (j < off) red[j] += red[j + off];
    __syncthreads();
  }
  float mean = red[0] * (1.f / HDIM); __syncthreads();
  float dv = v - mean;
  red[j] = dv * dv; __syncthreads();
  for (int off = HDIM / 2; off > 0; off >>= 1) {
    if (j < off) red[j] += red[j + off];
    __syncthreads();
  }
  float var = red[0] * (1.f / HDIM);
  float o = dv * rsqrtf(var + 1e-5f) * ng[j] + nb[j];
  o = fminf(fmaxf(o, 0.f), CLAMP_V);
  t[j] = o; __syncthreads();
  if (j < 5) {
    float s = bh[j];
    for (int k = 0; k < HDIM; ++k) s += t[k] * Wh[k * 5 + j];
    out[gidx * 5 + j] = s;
  }
}

// ---------------- launch ----------------
extern "C" void kernel_launch(void* const* d_in, const int* in_sizes, int n_in,
                              void* d_out, int out_size, void* d_ws, size_t ws_size,
                              hipStream_t stream) {
  (void)in_sizes; (void)n_in; (void)out_size; (void)ws_size;

  const float* x        = (const float*)d_in[0];
  const int*   ei       = (const int*)  d_in[1];
  const float* ea       = (const float*)d_in[2];
  const int*   batch    = (const int*)  d_in[3];
  const float* c1Wf     = (const float*)d_in[4];
  const float* c1bf     = (const float*)d_in[5];
  const float* c1Ws     = (const float*)d_in[6];
  const float* c1bs     = (const float*)d_in[7];
  const float* projW    = (const float*)d_in[8];
  const float* projb    = (const float*)d_in[9];
  const float* normg    = (const float*)d_in[10];
  const float* normb    = (const float*)d_in[11];
  const float* c2Wf     = (const float*)d_in[12];
  const float* c2bf     = (const float*)d_in[13];
  const float* c2Ws     = (const float*)d_in[14];
  const float* c2bs     = (const float*)d_in[15];
  const float* c3Wf     = (const float*)d_in[16];
  const float* c3bf     = (const float*)d_in[17];
  const float* c3Ws     = (const float*)d_in[18];
  const float* c3bs     = (const float*)d_in[19];
  const float* fc1W     = (const float*)d_in[20];
  const float* fc1b     = (const float*)d_in[21];
  const float* headW    = (const float*)d_in[22];
  const float* headb    = (const float*)d_in[23];
  float* out = (float*)d_out;

  // workspace layout (256B aligned segments)
  char* ws = (char*)d_ws;
  size_t off = 0;
  auto take = [&](size_t bytes) { char* p = ws + off; off = (off + bytes + 255) & ~(size_t)255; return p; };
  _Float16* h16    = (_Float16*)take((size_t)N_NODES * HDIM * 2);
  _Float16* ea16   = (_Float16*)take((size_t)N_EDGES * EDIM * 2);
  float*    hcur   = (float*)   take((size_t)N_NODES * HDIM * 4);
  float*    accb   = (float*)   take((size_t)N_NODES * HDIM * 4);
  float*    h1     = (float*)   take((size_t)N_NODES * NDIM * 4);
  _Float16* wpF    = (_Float16*)take((size_t)KTILES * NTILES * 32 * 16 * 2);
  _Float16* wpS    = (_Float16*)take((size_t)KTILES * NTILES * 32 * 16 * 2);
  float*    pooled = (float*)   take((size_t)N_GRAPH * HDIM * 4);
  float*    cnt    = (float*)   take((size_t)N_GRAPH * 4);

  const int NH = N_NODES * HDIM;
  const int numTiles = N_EDGES / TILE_E;       // 50000
  const int edgeBlocks = (numTiles + WAVES - 1) / WAVES;
  const int packThreads = KTILES * NTILES * 32;

  // conv1: h1 = x + scatter(gate)
  k_copy_f32<<<(N_NODES * NDIM + 255) / 256, 256, 0, stream>>>(h1, x, N_NODES * NDIM);
  k_conv1<<<(N_EDGES + 255) / 256, 256, 0, stream>>>(x, ei, ea, c1Wf, c1bf, c1Ws, c1bs, h1);

  // proj + LN + relu -> hcur/h16 ; ea -> f16 once
  k_proj_ln<<<N_NODES, HDIM, 0, stream>>>(h1, projW, projb, normg, normb, hcur, h16);
  k_f32_to_f16<<<(N_EDGES * EDIM + 255) / 256, 256, 0, stream>>>(ea, ea16, N_EDGES * EDIM);

  // conv2
  k_pack_w<<<(packThreads + 255) / 256, 256, 0, stream>>>(c2Wf, wpF);
  k_pack_w<<<(packThreads + 255) / 256, 256, 0, stream>>>(c2Ws, wpS);
  k_copy_f32<<<(NH + 255) / 256, 256, 0, stream>>>(accb, hcur, NH);   // residual init
  k_edge_wmma<<<edgeBlocks, WAVES * 32, 0, stream>>>(h16, ea16, ei, wpF, wpS,
                                                     c2bf, c2bs, accb, numTiles);
  k_post<<<(NH + 255) / 256, 256, 0, stream>>>(accb, hcur, h16, NH);

  // conv3
  k_pack_w<<<(packThreads + 255) / 256, 256, 0, stream>>>(c3Wf, wpF);
  k_pack_w<<<(packThreads + 255) / 256, 256, 0, stream>>>(c3Ws, wpS);
  k_copy_f32<<<(NH + 255) / 256, 256, 0, stream>>>(accb, hcur, NH);
  k_edge_wmma<<<edgeBlocks, WAVES * 32, 0, stream>>>(h16, ea16, ei, wpF, wpS,
                                                     c3bf, c3bs, accb, numTiles);
  k_post<<<(NH + 255) / 256, 256, 0, stream>>>(accb, hcur, h16, NH);

  // mean pool per graph + head
  hipMemsetAsync(pooled, 0, (size_t)N_GRAPH * HDIM * 4, stream);
  hipMemsetAsync(cnt, 0, (size_t)N_GRAPH * 4, stream);
  k_pool_cnt<<<(N_NODES + 255) / 256, 256, 0, stream>>>(batch, cnt);
  k_pool_sum<<<(NH + 255) / 256, 256, 0, stream>>>(hcur, batch, pooled);
  k_final<<<N_GRAPH, HDIM, 0, stream>>>(pooled, cnt, fc1W, fc1b, normg, normb,
                                        headW, headb, out);
}
